// _TripletMessage_56435870270129
// MI455X (gfx1250) — matrix operand order repaired and checked by reading, not compiled, not run on hardware
//
#include <hip/hip_runtime.h>
#include <math.h>

#define NEG_SLOPE 0.2f

typedef __attribute__((ext_vector_type(2)))  float  v2f;
typedef __attribute__((ext_vector_type(8)))  float  v8f;
typedef __attribute__((ext_vector_type(16))) __bf16 v16bf;

#if __has_builtin(__builtin_amdgcn_wmma_f32_16x16x4_f32)
#define HAVE_WMMA_F32 1
#else
#define HAVE_WMMA_F32 0
#endif

// ---------------------------------------------------------------------------
// GEMM: C[M,Nn] = A[M,K] @ B[K,Nn] (+ bias[col]).  One wave per block,
// each wave produces a 16x64 tile (4x v_wmma accumulators, A frag reused).
// M % 16 == 0, Nn % 64 == 0, K % 32 == 0 (K % 4 == 0 for f32 path).
// ---------------------------------------------------------------------------
__global__ void tm_gemm_wmma(const float* __restrict__ A, const float* __restrict__ B,
                             float* __restrict__ C, const float* __restrict__ bias,
                             int M, int Nn, int K) {
  const int lane = threadIdx.x & 31;
  const int row0 = blockIdx.x << 4;
  const int col0 = blockIdx.y << 6;
  const int nl = lane & 15;
  const int hi = lane >> 4;
  v8f acc[4] = {};
#if HAVE_WMMA_F32
  // fp32 WMMA: A 16x4 (2 VGPR: lanes0-15 K=k..k+1, lanes16-31 K=k+2..k+3)
  const int m  = row0 + nl;
  const int kb = hi * 2;
  for (int k = 0; k < K; k += 4) {
    v2f a;
    a.x = A[(size_t)m * K + (k + kb)];
    a.y = A[(size_t)m * K + (k + kb + 1)];
#pragma unroll
    for (int j = 0; j < 4; ++j) {
      const int col = col0 + j * 16 + nl;
      v2f b;
      b.x = B[(size_t)(k + kb) * Nn + col];
      b.y = B[(size_t)(k + kb + 1) * Nn + col];
      acc[j] = __builtin_amdgcn_wmma_f32_16x16x4_f32(false, a, false, b,
                                                     (short)0, acc[j], false, false);
    }
  }
#else
  // bf16 WMMA fallback (codegen-confirmed builtin), f32 accumulate.
  const int m   = row0 + nl;
  const int ka  = hi * 8;   // A: lanes0-15 hold K=0..7 & 16..23, lanes16-31 K=8..15 & 24..31
  const int kbB = hi * 8;   // B: lanes0-15 hold K=0..7 & 16..23, lanes16-31 K=8..15 & 24..31
  for (int k = 0; k < K; k += 32) {
    v16bf a;
#pragma unroll
    for (int i = 0; i < 8; ++i) {
      a[i]     = (__bf16)A[(size_t)m * K + (k + ka + i)];
      a[i + 8] = (__bf16)A[(size_t)m * K + (k + 16 + ka + i)];
    }
#pragma unroll
    for (int j = 0; j < 4; ++j) {
      const int col = col0 + j * 16 + nl;
      v16bf b;
#pragma unroll
      for (int i = 0; i < 16; ++i)
        b[i] = (__bf16)B[(size_t)(k + kbB + i + (i >= 8 ? 8 : 0)) * Nn + col];
      acc[j] = __builtin_amdgcn_wmma_f32_16x16x32_bf16(false, a, false, b,
                                                       (short)0, acc[j], false, false);
    }
  }
#endif
  // C/D layout: VGPR r -> M = r (+8 for lanes 16-31), N = lane&15
  const int mb = row0 + hi * 8;
#pragma unroll
  for (int j = 0; j < 4; ++j) {
    const int col = col0 + j * 16 + nl;
    const float bv = bias ? bias[col] : 0.0f;
#pragma unroll
    for (int r = 0; r < 8; ++r)
      C[(size_t)(mb + r) * Nn + col] = acc[j][r] + bv;
  }
}

// ---------------------------------------------------------------------------
__global__ void tm_init(float* __restrict__ aggr, float* __restrict__ mbuf,
                        float* __restrict__ denom, size_t aggr_n, int nh) {
  size_t t = (size_t)blockIdx.x * blockDim.x + threadIdx.x;
  if (t < aggr_n) aggr[t] = 0.0f;
  if (t < (size_t)nh) { mbuf[t] = -__builtin_inff(); denom[t] = 0.0f; }
}

// s0[n,h] = x_proj[n,h,:]·w[h,0,:]   s2[n,h] = x_proj[n,h,:]·w[h,2,:]
__global__ void tm_node_scores(const float* __restrict__ xp, const float* __restrict__ wt,
                               float* __restrict__ s0, float* __restrict__ s2, int NH) {
  int t = blockIdx.x * blockDim.x + threadIdx.x;
  if (t >= NH) return;
  int n = t / 3, h = t - 3 * n;
  const float* row = xp + (size_t)n * 192 + h * 64;
  const float* w0 = wt + h * 192;
  const float* w2 = wt + h * 192 + 128;
  float a0 = 0.f, a2 = 0.f;
#pragma unroll 4
  for (int c = 0; c < 64; ++c) { float v = row[c]; a0 += v * w0[c]; a2 += v * w2[c]; }
  s0[t] = a0; s2[t] = a2;
}

// we1[k*3+h] = sum_c W_edge[k, h*64+c] * w[h,1,c]  (folds edge proj into 16-dot)
__global__ void tm_we1(const float* __restrict__ We, const float* __restrict__ wt,
                       float* __restrict__ we1) {
  int t = threadIdx.x;
  if (t >= 48) return;
  int k = t / 3, h = t - 3 * k;
  const float* w1 = wt + h * 192 + 64;
  float s = 0.f;
#pragma unroll 4
  for (int c = 0; c < 64; ++c) s += We[k * 192 + h * 64 + c] * w1[c];
  we1[t] = s;
}

__device__ __forceinline__ void atomicMaxF32(float* addr, float val) {
  if (val >= 0.0f) atomicMax((int*)addr, __float_as_int(val));
  else             atomicMin((unsigned int*)addr, __float_as_uint(val));
}

// alpha[e,h] = leaky(s0[dst,h] + s2[src,h] + ea[e,:16]·we1[:,h]); atomic-max m[dst,h]
__global__ void tm_alpha(const int* __restrict__ ei, const float* __restrict__ ea,
                         const float* __restrict__ s0, const float* __restrict__ s2,
                         const float* __restrict__ we1, float* __restrict__ alpha,
                         float* __restrict__ mbuf, int E) {
  __shared__ float sw[48];
  if (threadIdx.x < 48) sw[threadIdx.x] = we1[threadIdx.x];
  __syncthreads();
  int e = blockIdx.x * blockDim.x + threadIdx.x;
  if (e >= E) return;
  const int src = ei[e];
  const int dst = ei[E + e];
  const float* eap = ea + (size_t)e * 16;
  float acc0 = 0.f, acc1 = 0.f, acc2 = 0.f;
#pragma unroll
  for (int k = 0; k < 16; ++k) {
    float v = eap[k];
    acc0 += v * sw[k * 3 + 0];
    acc1 += v * sw[k * 3 + 1];
    acc2 += v * sw[k * 3 + 2];
  }
  float acc[3] = {acc0, acc1, acc2};
#pragma unroll
  for (int h = 0; h < 3; ++h) {
    float a = acc[h] + s0[(size_t)dst * 3 + h] + s2[(size_t)src * 3 + h];
    a = (a >= 0.f) ? a : NEG_SLOPE * a;
    alpha[(size_t)e * 3 + h] = a;
    atomicMaxF32(&mbuf[(size_t)dst * 3 + h], a);
  }
}

// exp_a = exp(alpha - m[dst]); denom[dst,h] += exp_a  (in-place on alpha buffer)
__global__ void tm_expsum(const int* __restrict__ ei, float* __restrict__ alpha,
                          const float* __restrict__ mbuf, float* __restrict__ denom, int E) {
  int t = blockIdx.x * blockDim.x + threadIdx.x;
  if (t >= E * 3) return;
  int e = t / 3, h = t - 3 * e;
  const int dst = ei[E + e];
  float mm = mbuf[(size_t)dst * 3 + h];
  if (!(mm > -1e37f)) mm = 0.0f;   // mirrors reference finite-guard
  float v = __expf(alpha[t] - mm);
  alpha[t] = v;
  unsafeAtomicAdd(&denom[(size_t)dst * 3 + h], v);
}

__global__ void tm_invd(const float* __restrict__ denom, float* __restrict__ invd, int NH) {
  int t = blockIdx.x * blockDim.x + threadIdx.x;
  if (t < NH) invd[t] = 1.0f / (denom[t] + 1e-16f);
}

// Per edge: e_ij recomputed from edge_attr (W_edge in LDS), message scatter-add.
// 192 threads <-> 192 channels; block loops over EPB edges.
__global__ void tm_message(const int* __restrict__ ei, const float* __restrict__ ea,
                           const float* __restrict__ We, const float* __restrict__ xp,
                           const float* __restrict__ expa, const float* __restrict__ invd,
                           float* __restrict__ aggr, int E, int epb) {
  __shared__ float sWe[16 * 192];
  __shared__ float sea[16];
  __shared__ float scf[3];
  const int i = threadIdx.x;            // channel 0..191
  const int h = i >> 6;
  for (int t = i; t < 16 * 192; t += blockDim.x) sWe[t] = We[t];
  __syncthreads();
  const int e0 = blockIdx.x * epb;
  const int e1 = min(e0 + epb, E);
  for (int e = e0; e < e1; ++e) {
    const int src = ei[e];
    const int dst = ei[E + e];
    if (i < 16) sea[i] = ea[(size_t)e * 16 + i];
    if (i < 3)  scf[i] = expa[(size_t)e * 3 + i] * invd[(size_t)dst * 3 + i];
    __syncthreads();
    float eij = 0.f;
#pragma unroll
    for (int k = 0; k < 16; ++k) eij += sea[k] * sWe[k * 192 + i];
    const float msg = scf[h] * eij * xp[(size_t)src * 192 + i];
    unsafeAtomicAdd(&aggr[(size_t)dst * 192 + i], msg);
    __syncthreads();
  }
}

// ---------------------------------------------------------------------------
extern "C" void kernel_launch(void* const* d_in, const int* in_sizes, int n_in,
                              void* d_out, int out_size, void* d_ws, size_t ws_size,
                              hipStream_t stream) {
  const float* x    = (const float*)d_in[0];
  const int*   ei   = (const int*)d_in[1];      // edge_index [2,E] (JAX default int32)
  const float* ea   = (const float*)d_in[2];    // [E,16]
  const float* Wn   = (const float*)d_in[3];    // [64,192]
  const float* We   = (const float*)d_in[4];    // [16,192]
  const float* Wt   = (const float*)d_in[5];    // [3,3,64] flat
  const float* Wsc  = (const float*)d_in[6];    // [192,64]
  const float* bias = (const float*)d_in[7];    // [64]
  float* out = (float*)d_out;

  const int N = in_sizes[0] / 64;   // 50000
  const int E = in_sizes[1] / 2;    // 800000

  float* p = (float*)d_ws;
  float* xp    = p; p += (size_t)N * 192;
  float* aggr  = p; p += (size_t)N * 192;
  float* s0    = p; p += (size_t)N * 3;
  float* s2    = p; p += (size_t)N * 3;
  float* mbuf  = p; p += (size_t)N * 3;
  float* denom = p; p += (size_t)N * 3;
  float* invd  = p; p += (size_t)N * 3;
  float* expa  = p; p += (size_t)E * 3;
  float* we1   = p; p += 64;

  const size_t aggr_n = (size_t)N * 192;
  tm_init<<<dim3((unsigned)((aggr_n + 255) / 256)), 256, 0, stream>>>(aggr, mbuf, denom,
                                                                      aggr_n, N * 3);
  // x_proj = x @ W_node  (50000x64 @ 64x192)
  tm_gemm_wmma<<<dim3(N / 16, 192 / 64), 32, 0, stream>>>(x, Wn, xp, nullptr, N, 192, 64);
  tm_node_scores<<<(N * 3 + 255) / 256, 256, 0, stream>>>(xp, Wt, s0, s2, N * 3);
  tm_we1<<<1, 64, 0, stream>>>(We, Wt, we1);
  tm_alpha<<<(E + 255) / 256, 256, 0, stream>>>(ei, ea, s0, s2, we1, expa, mbuf, E);
  tm_expsum<<<(E * 3 + 255) / 256, 256, 0, stream>>>(ei, expa, mbuf, denom, E);
  tm_invd<<<(N * 3 + 255) / 256, 256, 0, stream>>>(denom, invd, N * 3);
  const int EPB = 8;
  tm_message<<<(E + EPB - 1) / EPB, 192, 0, stream>>>(ei, ea, We, xp, expa, invd, aggr, E, EPB);
  // out = aggr @ W_scale + bias  (50000x192 @ 192x64)
  tm_gemm_wmma<<<dim3(N / 16, 64 / 64), 32, 0, stream>>>(aggr, Wsc, out, bias, N, 64, 192);
}
